// LinearAttention_37864431681775
// MI455X (gfx1250) — matrix operand order repaired and checked by reading, not compiled, not run on hardware
//
#include <hip/hip_runtime.h>
#include <hip/hip_bf16.h>
#include <stdint.h>

// ---------------- problem constants (fixed by the reference) ----------------
#define BB_   4
#define S_    4096
#define H_    2048
#define KNH   16
#define HD_   128
#define FD_   64
#define BS_   (BB_ * S_)

// ---------------- types ----------------
typedef __bf16 bf16_t;
typedef __attribute__((ext_vector_type(16))) __bf16 v16bf;
typedef __attribute__((ext_vector_type(8)))  __bf16 v8bf;
typedef __attribute__((ext_vector_type(8)))  float  v8f;
typedef __attribute__((ext_vector_type(4)))  float  v4f;

enum {
    EPI_BIAS_BF16  = 0,  // C(bf16) = acc + bias[n]
    EPI_BIAS_F32   = 1,  // C(f32)  = acc + bias[n]
    EPI_RELU_BF16  = 2,  // C(bf16) = max(acc,0)
    EPI_RELU_MASK_T= 3,  // C(bf16)[n][m] = max(acc,0) * mask[m]   (transposed store)
    EPI_ATOMIC_F32 = 4,  // atomicAdd(C(f32)[m][n], acc)           (split-K partials)
    EPI_DIV_F32    = 5   // C(f32) = acc / (qk_sum[m] + 1e-8)      (rcp from prologue)
};

// ---------------- generic WMMA-bf16 tiled GEMM, double-buffered LDS ----------------
// Computes C[M,N] += A[M,K] * B^T where B is addressed as B[n*sBn + k*sBk].
// A is K-contiguous with row stride lda. Batch (grid.z) base offsets:
//   base = (bz/KNH)*S1 + (bz%KNH)*S2 for each operand.
template <typename AT, int BM, int BN, bool B_NFAST, bool SPLITK, int EPI>
__global__ __launch_bounds__(256)
void gemm_wmma(const void* __restrict__ Av, long lda, long aS1, long aS2,
               const bf16_t* __restrict__ Bg, long sBn, long sBk, long bS1, long bS2,
               void* __restrict__ Cv, long ldc, long cS1, long cS2,
               int K,
               const float* __restrict__ aux, long auxS1,
               const float* __restrict__ mask, long maskS)
{
    constexpr int BK      = 32;            // one wmma_f32_16x16x32_bf16 K-step
    constexpr int WAVES_N = BN / 32;       // 2 or 4
    constexpr int WAVES_M = 8 / WAVES_N;   // 4 or 2
    constexpr int ITN     = 2;
    constexpr int ITM     = (BM / WAVES_M) / 16;
    constexpr int PTA     = BM * BK / 8 / 256;   // 8-elem chunks per thread (A)
    constexpr int PTB     = BN * BK / 8 / 256;   // 8-elem chunks per thread (B)

    __shared__ alignas(64) bf16_t As[2][BM][BK];
    __shared__ alignas(64) bf16_t Bs[2][BN][BK];
    __shared__ float sRcp[(EPI == EPI_DIV_F32) ? BM : 1];

    const int tid  = threadIdx.x;
    const int lane = tid & 31;
    const int wave = tid >> 5;
    const int wn   = wave % WAVES_N;
    const int wm   = wave / WAVES_N;
    const int half = lane >> 4;
    const int l16  = lane & 15;

    const int bz = blockIdx.z;
    const int bb = bz / KNH;
    const int hh = bz % KNH;

    const AT*     Ab = (const AT*)Av + (long)bb * aS1 + (long)hh * aS2;
    const bf16_t* Bb = Bg            + (long)bb * bS1 + (long)hh * bS2;
    bf16_t* C16 = (bf16_t*)Cv + (long)bb * cS1 + (long)hh * cS2;
    float*  C32 = (float*)Cv  + (long)bb * cS1 + (long)hh * cS2;
    const float* auxb  = aux  ? aux  + (long)bz * auxS1 : nullptr;
    const float* maskb = mask ? mask + (long)bb * maskS : nullptr;

    int bm = blockIdx.y;
    int bn;
    if constexpr (SPLITK) {
        bn = 0;
        const long k0b = (long)blockIdx.x * (long)K;
        Ab += k0b;                 // A's K stride is 1
        Bb += k0b * sBk;
    } else {
        bn = blockIdx.x;
    }
    const long rowA0 = (long)bm * BM;

    // prologue for the fused divide: rcp(q' . k_sum + 1e-8) per output row
    if constexpr (EPI == EPI_DIV_F32) {
        for (int r = tid; r < BM; r += 256) {
            const AT* ap = Ab + (rowA0 + r) * lda;
            float s = 0.f;
            for (int k = 0; k < K; ++k) s += (float)ap[k] * auxb[k];
            sRcp[r] = 1.0f / (s + 1e-8f);
        }
        __syncthreads();
    }

    // per-thread staging coordinates (loop-invariant)
    int rA[PTA], kA[PTA];
#pragma unroll
    for (int c = 0; c < PTA; ++c) {
        const int ch = tid + c * 256;
        rA[c] = ch / (BK / 8);
        kA[c] = (ch % (BK / 8)) * 8;
    }
    int nB[PTB], kB[PTB];
#pragma unroll
    for (int c = 0; c < PTB; ++c) {
        const int ch = tid + c * 256;
        if constexpr (!B_NFAST) { nB[c] = ch / (BK / 8); kB[c] = (ch % (BK / 8)) * 8; }
        else                    { nB[c] = (ch % (BN / 8)) * 8; kB[c] = ch / (BN / 8); }
    }

    // register staging for the in-flight tile
    v4f  aF[PTA][2];
    v8bf aH[PTA];
    v8bf bR[PTB];

    auto loadG = [&](int kb) {
        const int k0 = kb * BK;
#pragma unroll
        for (int c = 0; c < PTA; ++c) {
            const AT* src = Ab + (rowA0 + rA[c]) * lda + (k0 + kA[c]);
            __builtin_prefetch(src + BK, 0, 1);      // hint tile kb+1
            if constexpr (sizeof(AT) == 4) {
                aF[c][0] = *(const v4f*)src;
                aF[c][1] = *(const v4f*)(src + 4);
            } else {
                aH[c] = *(const v8bf*)src;
            }
        }
#pragma unroll
        for (int c = 0; c < PTB; ++c) {
            const bf16_t* src;
            if constexpr (!B_NFAST)
                src = Bb + (long)(bn * BN + nB[c]) * sBn + (k0 + kB[c]);
            else
                src = Bb + (long)(k0 + kB[c]) * sBk + (bn * BN + nB[c]);
            bR[c] = *(const v8bf*)src;
        }
    };
    auto storeL = [&](int buf) {
#pragma unroll
        for (int c = 0; c < PTA; ++c) {
            v8bf o;
            if constexpr (sizeof(AT) == 4) {
#pragma unroll
                for (int e = 0; e < 4; ++e) { o[e] = (bf16_t)aF[c][0][e]; o[4 + e] = (bf16_t)aF[c][1][e]; }
            } else {
                o = aH[c];
            }
            *(v8bf*)&As[buf][rA[c]][kA[c]] = o;
        }
#pragma unroll
        for (int c = 0; c < PTB; ++c) {
            if constexpr (!B_NFAST) {
                *(v8bf*)&Bs[buf][nB[c]][kB[c]] = bR[c];
            } else {
#pragma unroll
                for (int e = 0; e < 8; ++e) Bs[buf][nB[c] + e][kB[c]] = bR[c][e];
            }
        }
    };

    v8f acc[ITM][ITN];
#pragma unroll
    for (int i = 0; i < ITM; ++i)
#pragma unroll
        for (int j = 0; j < ITN; ++j)
#pragma unroll
            for (int e = 0; e < 8; ++e) acc[i][j][e] = 0.0f;

    const int kSteps = K / BK;

    // prime the pipeline
    loadG(0);
    storeL(0);
    __syncthreads();

    for (int kb = 0; kb < kSteps; ++kb) {
        const int cur = kb & 1;
        if (kb + 1 < kSteps) loadG(kb + 1);   // global loads overlap with WMMA below

        // ---- fragments (wave32 layouts per CDNA5 ISA 7.12.2) ----
        v16bf afr[ITM], bfr[ITN];
#pragma unroll
        for (int i = 0; i < ITM; ++i) {
            const int r = wm * (ITM * 16) + i * 16 + l16;
            union { v16bf v; v8bf h[2]; } u;
            u.h[0] = *(const v8bf*)&As[cur][r][half * 8];        // K = half*8 .. +7
            u.h[1] = *(const v8bf*)&As[cur][r][16 + half * 8];   // K = 16+half*8 .. +7
            afr[i] = u.v;
        }
#pragma unroll
        for (int j = 0; j < ITN; ++j) {
            const int n = wn * (ITN * 16) + j * 16 + l16;
            union { v16bf v; v8bf h[2]; } w;
            w.h[0] = *(const v8bf*)&Bs[cur][n][half * 8];        // K = half*8 .. +7
            w.h[1] = *(const v8bf*)&Bs[cur][n][16 + half * 8];   // K = 16+half*8 .. +7
            bfr[j] = w.v;
        }
#pragma unroll
        for (int i = 0; i < ITM; ++i)
#pragma unroll
            for (int j = 0; j < ITN; ++j)
                acc[i][j] = __builtin_amdgcn_wmma_f32_16x16x32_bf16(
                    false, afr[i], false, bfr[j], (short)0, acc[i][j], false, false);

        if (kb + 1 < kSteps) {
            storeL((kb + 1) & 1);   // fill the other buffer
            __syncthreads();        // single barrier per K-step
        }
    }

    // ---- epilogue: D layout = 8 f32 VGPRs, M = r + 8*(lane>=16), N = lane&15 ----
#pragma unroll
    for (int i = 0; i < ITM; ++i) {
        const int mBase = wm * (ITM * 16) + i * 16 + half * 8;
#pragma unroll
        for (int j = 0; j < ITN; ++j) {
            const int nLoc = wn * (ITN * 16) + j * 16 + l16;
            const long ng  = (long)bn * BN + nLoc;
#pragma unroll
            for (int r = 0; r < 8; ++r) {
                const int  mLoc = mBase + r;
                const long mg   = rowA0 + mLoc;
                float v = acc[i][j][r];
                if constexpr (EPI == EPI_BIAS_BF16) {
                    v += auxb[ng];
                    C16[mg * ldc + ng] = (bf16_t)v;
                } else if constexpr (EPI == EPI_BIAS_F32) {
                    v += auxb[ng];
                    C32[mg * ldc + ng] = v;
                } else if constexpr (EPI == EPI_RELU_BF16) {
                    v = fmaxf(v, 0.0f);
                    C16[mg * ldc + ng] = (bf16_t)v;
                } else if constexpr (EPI == EPI_RELU_MASK_T) {
                    v = fmaxf(v, 0.0f) * maskb[mg];
                    C16[ng * ldc + mg] = (bf16_t)v;      // transposed: [FD][S]
                } else if constexpr (EPI == EPI_ATOMIC_F32) {
                    atomicAdd(&C32[mg * ldc + ng], v);
                } else { // EPI_DIV_F32
                    v *= sRcp[mLoc];
                    C32[mg * ldc + ng] = v;
                }
            }
        }
    }
}

// ---------------- helpers ----------------
__global__ __launch_bounds__(256) void cvt_f32_bf16(const float* __restrict__ s,
                                                    bf16_t* __restrict__ d, long n) {
    long i = (long)blockIdx.x * 256 + threadIdx.x;
    const long st = (long)gridDim.x * 256;
    for (; i < n; i += st) d[i] = (bf16_t)s[i];
}

__global__ __launch_bounds__(256) void zero_f32(float* __restrict__ p, long n) {
    long i = (long)blockIdx.x * 256 + threadIdx.x;
    const long st = (long)gridDim.x * 256;
    for (; i < n; i += st) p[i] = 0.0f;
}

// one block per (b,h,f) row: k_sum = sum_n k'[b,h,f,n]  (k' is [.., FD, S], contiguous)
__global__ __launch_bounds__(256) void ksum_kernel(const bf16_t* __restrict__ kT,
                                                   float* __restrict__ ksum) {
    __shared__ float red[256];
    const long row = blockIdx.x;
    const bf16_t* p = kT + row * (long)S_;
    float s = 0.f;
    for (int i = threadIdx.x; i < S_; i += 256) s += (float)p[i];
    red[threadIdx.x] = s;
    __syncthreads();
    for (int o = 128; o > 0; o >>= 1) {
        if (threadIdx.x < o) red[threadIdx.x] += red[threadIdx.x + o];
        __syncthreads();
    }
    if (threadIdx.x == 0) ksum[row] = red[0];
}

// ---------------- host orchestration ----------------
extern "C" void kernel_launch(void* const* d_in, const int* in_sizes, int n_in,
                              void* d_out, int out_size, void* d_ws, size_t ws_size,
                              hipStream_t stream) {
    (void)in_sizes; (void)n_in; (void)out_size; (void)ws_size;
    const float* query = (const float*)d_in[0];
    const float* key   = (const float*)d_in[1];
    const float* value = (const float*)d_in[2];
    const float* amask = (const float*)d_in[3];
    const float* Wq    = (const float*)d_in[4];
    const float* bq    = (const float*)d_in[5];
    const float* Wk    = (const float*)d_in[6];
    const float* bk    = (const float*)d_in[7];
    const float* Wv    = (const float*)d_in[8];
    const float* bv    = (const float*)d_in[9];
    const float* Wqf   = (const float*)d_in[10];
    const float* Wkf   = (const float*)d_in[11];
    const float* Wo    = (const float*)d_in[12];
    const float* bo    = (const float*)d_in[13];
    float* out = (float*)d_out;

    char* ws = (char*)d_ws;
    size_t off = 0;
    auto take = [&](size_t bytes) -> char* {
        char* p = ws + off;
        off += (bytes + 255) & ~(size_t)255;
        return p;
    };
    bf16_t* wq  = (bf16_t*)take((size_t)H_ * H_ * 2);
    bf16_t* wk  = (bf16_t*)take((size_t)H_ * H_ * 2);
    bf16_t* wv  = (bf16_t*)take((size_t)H_ * H_ * 2);
    bf16_t* wo  = (bf16_t*)take((size_t)H_ * H_ * 2);
    bf16_t* wqf = (bf16_t*)take((size_t)FD_ * HD_ * 2);
    bf16_t* wkf = (bf16_t*)take((size_t)FD_ * HD_ * 2);
    bf16_t* qh  = (bf16_t*)take((size_t)BS_ * H_ * 2);
    bf16_t* kh  = (bf16_t*)take((size_t)BS_ * H_ * 2);
    bf16_t* vh  = (bf16_t*)take((size_t)BS_ * H_ * 2);
    bf16_t* qp  = (bf16_t*)take((size_t)BB_ * KNH * S_ * FD_ * 2);   // [B,NH,S,FD]
    bf16_t* kpT = (bf16_t*)take((size_t)BB_ * KNH * FD_ * S_ * 2);   // [B,NH,FD,S]
    float*  kvf = (float*) take((size_t)BB_ * KNH * FD_ * HD_ * 4);  // [B,NH,FD,HD]
    bf16_t* kvb = (bf16_t*)take((size_t)BB_ * KNH * FD_ * HD_ * 2);
    float*  ksm = (float*) take((size_t)BB_ * KNH * FD_ * 4);        // [B,NH,FD]
    float*  att = (float*) take((size_t)BS_ * H_ * 4);               // [BS,H]

    const dim3 blk(256);
    auto cvt = [&](const float* s, bf16_t* d, long n) {
        int g = (int)((n + 255) / 256); if (g > 4096) g = 4096;
        cvt_f32_bf16<<<g, blk, 0, stream>>>(s, d, n);
    };
    // 1) weights -> bf16
    cvt(Wq, wq, (long)H_ * H_);  cvt(Wk, wk, (long)H_ * H_);
    cvt(Wv, wv, (long)H_ * H_);  cvt(Wo, wo, (long)H_ * H_);
    cvt(Wqf, wqf, (long)FD_ * HD_);  cvt(Wkf, wkf, (long)FD_ * HD_);

    // 2) projections: [BS,H] = x @ W^T + b  (bf16 out)
    {
        dim3 grid(H_ / 128, BS_ / 128, 1);
        gemm_wmma<float, 128, 128, false, false, EPI_BIAS_BF16><<<grid, blk, 0, stream>>>(
            query, H_, 0, 0,  wq, H_, 1, 0, 0,  qh, H_, 0, 0,  H_, bq, 0, nullptr, 0);
        gemm_wmma<float, 128, 128, false, false, EPI_BIAS_BF16><<<grid, blk, 0, stream>>>(
            key,   H_, 0, 0,  wk, H_, 1, 0, 0,  kh, H_, 0, 0,  H_, bk, 0, nullptr, 0);
        gemm_wmma<float, 128, 128, false, false, EPI_BIAS_BF16><<<grid, blk, 0, stream>>>(
            value, H_, 0, 0,  wv, H_, 1, 0, 0,  vh, H_, 0, 0,  H_, bv, 0, nullptr, 0);
    }

    // 3) feature maps per (b,h): q' = relu(qh @ Wqf^T); k' = relu(kh @ Wkf^T)*mask (transposed out)
    {
        dim3 grid(FD_ / 64, S_ / 128, BB_ * KNH);
        gemm_wmma<bf16_t, 128, 64, false, false, EPI_RELU_BF16><<<grid, blk, 0, stream>>>(
            qh, H_, (long)S_ * H_, HD_,  wqf, HD_, 1, 0, 0,
            qp, FD_, (long)KNH * S_ * FD_, (long)S_ * FD_,  HD_, nullptr, 0, nullptr, 0);
        gemm_wmma<bf16_t, 128, 64, false, false, EPI_RELU_MASK_T><<<grid, blk, 0, stream>>>(
            kh, H_, (long)S_ * H_, HD_,  wkf, HD_, 1, 0, 0,
            kpT, S_, (long)KNH * FD_ * S_, (long)FD_ * S_,  HD_, nullptr, 0, amask, S_);
    }

    // 4) k_sum + kv = k'^T @ v (split-K, atomic f32 accumulation) + bf16 convert
    {
        const long nkv = (long)BB_ * KNH * FD_ * HD_;
        zero_f32<<<(int)((nkv + 255) / 256), blk, 0, stream>>>(kvf, nkv);
        ksum_kernel<<<BB_ * KNH * FD_, blk, 0, stream>>>(kpT, ksm);
        dim3 grid(8 /*K-split*/, 1, BB_ * KNH);
        gemm_wmma<bf16_t, 64, 128, true, true, EPI_ATOMIC_F32><<<grid, blk, 0, stream>>>(
            kpT, S_, (long)KNH * FD_ * S_, (long)FD_ * S_,
            vh, 1, H_, (long)S_ * H_, HD_,
            kvf, HD_, (long)KNH * FD_ * HD_, (long)FD_ * HD_,
            S_ / 8, nullptr, 0, nullptr, 0);
        cvt(kvf, kvb, nkv);
    }

    // 5) qkv = q' @ kv, fused divide by (q'.k_sum + 1e-8) -> attn f32 [BS,H]
    {
        dim3 grid(HD_ / 128, S_ / 128, BB_ * KNH);
        gemm_wmma<bf16_t, 128, 128, true, false, EPI_DIV_F32><<<grid, blk, 0, stream>>>(
            qp, FD_, (long)KNH * S_ * FD_, (long)S_ * FD_,
            kvb, 1, HD_, (long)KNH * FD_ * HD_, (long)FD_ * HD_,
            att, H_, (long)S_ * H_, HD_,
            FD_, ksm, FD_, nullptr, 0);
    }

    // 6) output projection: out = attn @ Wo^T + bo  (f32)
    {
        dim3 grid(H_ / 128, BS_ / 128, 1);
        gemm_wmma<float, 128, 128, false, false, EPI_BIAS_F32><<<grid, blk, 0, stream>>>(
            att, H_, 0, 0,  wo, H_, 1, 0, 0,  out, H_, 0, 0,  H_, bo, 0, nullptr, 0);
    }
}
